// SimpleDiagSSM_13915694039802
// MI455X (gfx1250) — matrix operand order repaired and checked by reading, not compiled, not run on hardware
//
#include <hip/hip_runtime.h>
#include <hip/hip_bf16.h>
#include <math.h>

// ---------------- problem constants ----------------
#define D_MODEL 1024
#define D_INNER 2048
#define D_CONV  4
#define BATCH   4
#define TLEN    2048
#define MROWS   (BATCH * TLEN)     // 8192 rows for all GEMMs
#define NCHUNK  32                 // scan chunks over T
#define CHUNK   (TLEN / NCHUNK)    // 64

typedef unsigned short u16;
typedef __attribute__((ext_vector_type(16))) __bf16 v16bf;
typedef __attribute__((ext_vector_type(8)))  float  v8f;

// ---------------- small helpers ----------------
__device__ __forceinline__ u16 f32_to_bf16_rne(float f) {
  union { float f; unsigned int u; } c; c.f = f;
  unsigned int u = c.u;
  unsigned int r = u + 0x7FFFu + ((u >> 16) & 1u);
  if ((u & 0x7F800000u) == 0x7F800000u) r = u;   // inf/nan passthrough
  return (u16)(r >> 16);
}

__device__ __forceinline__ float silu_f(float x) {
  return x * (1.0f / (1.0f + __expf(-x)));
}

__device__ __forceinline__ v8f wmma_bf16(v16bf a, v16bf b, v8f c) {
  // (neg_a, A, neg_b, B, c_mod, C, reuse_a, reuse_b)
  return __builtin_amdgcn_wmma_f32_16x16x32_bf16(false, a, false, b, (short)0, c,
                                                 false, false);
}

// async global -> LDS copy of 16 bytes (CDNA5 GLOBAL_LOAD_ASYNC_TO_LDS_B128),
// tracked by ASYNCcnt. lds = wave-relative LDS byte offset (low 32 bits of the
// generic shared-space address per the LDS aperture rule).
__device__ __forceinline__ void async_copy_b128(unsigned lds, const void* gaddr) {
  asm volatile("global_load_async_to_lds_b128 %0, %1, off"
               :: "v"(lds), "v"(gaddr) : "memory");
}

__device__ __forceinline__ unsigned lds_off(const void* p) {
  return (unsigned)(uintptr_t)p;
}

// ---------------- f32 -> bf16 conversion ----------------
__global__ __launch_bounds__(256) void cvt_f32_bf16_kernel(
    const float* __restrict__ src, u16* __restrict__ dst, int n) {
  int i = blockIdx.x * blockDim.x + threadIdx.x;
  if (i < n) dst[i] = f32_to_bf16_rne(src[i]);
}

// ---------------- WMMA bf16 GEMM: C[M,N] = A[M,K] * W[N,K]^T + bias ----------------
// EPI 0: C = acc + bias
// EPI 2: C = softplus(acc + bias) * aux[m*N + n]   (dt * x_conv fusion)
#define BM 128
#define BN 128
#define BK 32
#define LDSW 40   // ushorts per LDS row (32 + 8 pad); 80B, 16B-aligned, conflict-free

template <int EPI>
__global__ __launch_bounds__(256) void gemm_bf16_wmma_kernel(
    const u16* __restrict__ A, const u16* __restrict__ W,
    const float* __restrict__ bias, float* __restrict__ C,
    const float* __restrict__ aux, int M, int N, int K) {
  __shared__ u16 As[2][BM * LDSW];
  __shared__ u16 Bs[2][BN * LDSW];

  const int tid  = threadIdx.x;
  const int lane = tid & 31;
  const int wave = tid >> 5;
  const int wm   = wave & 3;    // wave M offset = 32*wm
  const int wn   = wave >> 2;   // wave N offset = 64*wn
  const int l15  = lane & 15;
  const int lhi  = lane >> 4;

  const int m0 = blockIdx.y * BM;
  const int n0 = blockIdx.x * BN;

  // staging pattern: 4 threads per row, 8 bf16 (16B) each; 2 row passes
  const int lrow = tid >> 2;          // 0..63
  const int lcol = (tid & 3) * 8;     // 0,8,16,24

  // per-thread global source pointers (advance by k0 in the loop)
  const u16* gA0 = &A[(size_t)(m0 + lrow     ) * K + lcol];
  const u16* gA1 = &A[(size_t)(m0 + lrow + 64) * K + lcol];
  const u16* gB0 = &W[(size_t)(n0 + lrow     ) * K + lcol];
  const u16* gB1 = &W[(size_t)(n0 + lrow + 64) * K + lcol];

  // per-thread LDS destination offsets for both buffers
  unsigned dA0[2], dA1[2], dB0[2], dB1[2];
#pragma unroll
  for (int b = 0; b < 2; ++b) {
    dA0[b] = lds_off(&As[b][(lrow     ) * LDSW + lcol]);
    dA1[b] = lds_off(&As[b][(lrow + 64) * LDSW + lcol]);
    dB0[b] = lds_off(&Bs[b][(lrow     ) * LDSW + lcol]);
    dB1[b] = lds_off(&Bs[b][(lrow + 64) * LDSW + lcol]);
  }

  v8f acc[2][4];
#pragma unroll
  for (int i = 0; i < 2; ++i)
#pragma unroll
    for (int j = 0; j < 4; ++j)
#pragma unroll
      for (int r = 0; r < 8; ++r) acc[i][j][r] = 0.0f;

  union FragU { uint4 q[2]; v16bf v; };

  const int nk = K / BK;

  // prologue: async-stage tile 0 into buffer 0 (4 async b128 per thread)
  async_copy_b128(dA0[0], gA0);
  async_copy_b128(dA1[0], gA1);
  async_copy_b128(dB0[0], gB0);
  async_copy_b128(dB1[0], gB1);

  for (int kt = 0; kt < nk; ++kt) {
    const int cur = kt & 1;
    __syncthreads();   // all reads of the buffer we are about to overwrite done
    if (kt + 1 < nk) {
      const int nxt = cur ^ 1;
      const int k1  = (kt + 1) * BK;
      async_copy_b128(dA0[nxt], gA0 + k1);
      async_copy_b128(dA1[nxt], gA1 + k1);
      async_copy_b128(dB0[nxt], gB0 + k1);
      async_copy_b128(dB1[nxt], gB1 + k1);
      // async ops complete in order: <=4 outstanding => tile kt has landed
      asm volatile("s_wait_asynccnt 0x4" ::: "memory");
    } else {
      asm volatile("s_wait_asynccnt 0x0" ::: "memory");
    }
    __syncthreads();   // tile kt visible to every wave

    // A fragments: lane holds row m = base + lane%16;
    // halves 0..7 = K[lhi*8 .. +7], halves 8..15 = K[16 + lhi*8 .. +7]
    FragU fa[2];
#pragma unroll
    for (int i = 0; i < 2; ++i) {
      int row = wm * 32 + i * 16 + l15;
      int kh  = lhi * 8;
      fa[i].q[0] = *(const uint4*)&As[cur][row * LDSW + kh];
      fa[i].q[1] = *(const uint4*)&As[cur][row * LDSW + kh + 16];
    }
    // B fragments: lane holds col n = base + lane%16;
    // halves 0..7 = K[lhi*8 .. +7], halves 8..15 = K[16 + lhi*8 .. +7]
    FragU fb[4];
#pragma unroll
    for (int j = 0; j < 4; ++j) {
      int row = wn * 64 + j * 16 + l15;
      int kb  = lhi * 8;
      fb[j].q[0] = *(const uint4*)&Bs[cur][row * LDSW + kb];
      fb[j].q[1] = *(const uint4*)&Bs[cur][row * LDSW + kb + 16];
    }
#pragma unroll
    for (int i = 0; i < 2; ++i)
#pragma unroll
      for (int j = 0; j < 4; ++j)
        acc[i][j] = wmma_bf16(fa[i].v, fb[j].v, acc[i][j]);
  }

  // epilogue: C/D layout -> element r: m = base + r + 8*(lane>=16), n = base + lane%16
#pragma unroll
  for (int i = 0; i < 2; ++i) {
#pragma unroll
    for (int j = 0; j < 4; ++j) {
      int mbase = m0 + wm * 32 + i * 16 + lhi * 8;
      int n     = n0 + wn * 64 + j * 16 + l15;
      float bn  = bias[n];
#pragma unroll
      for (int r = 0; r < 8; ++r) {
        int m   = mbase + r;
        float v = acc[i][j][r] + bn;
        if (EPI == 2) {
          float dt = (v > 20.0f) ? v : log1pf(__expf(v));
          v = dt * aux[(size_t)m * N + n];
        }
        C[(size_t)m * N + n] = v;
      }
    }
  }
}

// ---------------- causal depthwise conv (k=4) + SiLU ----------------
// reads x_branch = xz[:, 0:D_INNER]; writes f32 and bf16 copies of x_conv
__global__ __launch_bounds__(256) void conv_silu_kernel(
    const float* __restrict__ xz, const float* __restrict__ Wc,
    const float* __restrict__ bc, float* __restrict__ xcf,
    u16* __restrict__ xcb) {
  int idx = blockIdx.x * blockDim.x + threadIdx.x;
  if (idx >= MROWS * D_INNER) return;
  int c = idx & (D_INNER - 1);
  int m = idx >> 11;          // / D_INNER
  int t = m & (TLEN - 1);

  float s = bc[c];
#pragma unroll
  for (int j = 0; j < D_CONV; ++j) {
    int tt = t - (D_CONV - 1) + j;
    if (tt >= 0)
      s += Wc[c * D_CONV + j] *
           xz[(size_t)(m - (D_CONV - 1) + j) * (2 * D_INNER) + c];
  }
  float v = silu_f(s);
  xcf[idx] = v;
  xcb[idx] = f32_to_bf16_rne(v);
}

// ---------------- chunked diagonal scan ----------------
// pass 1: local scan within each chunk of 64 steps; vals overwritten in place
// with the local prefix h; chunk-final h written to carry[b][chunk][c]
__global__ __launch_bounds__(256) void scan_local_kernel(
    float* __restrict__ vals, float* __restrict__ carry,
    const float* __restrict__ decay) {
  int idx = blockIdx.x * blockDim.x + threadIdx.x;       // (b,chunk,c)
  if (idx >= BATCH * NCHUNK * D_INNER) return;
  int c  = idx & (D_INNER - 1);
  int r  = idx >> 11;
  int ch = r & (NCHUNK - 1);
  int b  = r >> 5;
  float d = decay[c];
  float h = 0.0f;
  size_t base = ((size_t)b * TLEN + (size_t)ch * CHUNK) * D_INNER + c;
  for (int i = 0; i < CHUNK; ++i) {
    size_t p = base + (size_t)i * D_INNER;
    h = d * h + vals[p];
    vals[p] = h;
  }
  carry[((size_t)b * NCHUNK + ch) * D_INNER + c] = h;
}

// pass 2: exclusive prefix of chunk carries per (b,c): e[0]=0, e[j]=f[j-1]+d^64*e[j-1]
__global__ __launch_bounds__(256) void scan_carry_kernel(
    float* __restrict__ carry, const float* __restrict__ decay) {
  int idx = blockIdx.x * blockDim.x + threadIdx.x;       // (b,c)
  if (idx >= BATCH * D_INNER) return;
  int c = idx & (D_INNER - 1);
  int b = idx >> 11;
  float d = decay[c];
  float dC = d;                       // d^CHUNK via 6 squarings (CHUNK=64)
#pragma unroll
  for (int s = 0; s < 6; ++s) dC *= dC;
  float e = 0.0f;
  for (int ch = 0; ch < NCHUNK; ++ch) {
    size_t p = ((size_t)b * NCHUNK + ch) * D_INNER + c;
    float f = carry[p];
    carry[p] = e;
    e = f + dC * e;
  }
}

// pass 3: y = (h_local + d^(i+1) * carry_in) * silu(z); write bf16 for GEMM3
__global__ __launch_bounds__(256) void scan_fixup_gate_kernel(
    const float* __restrict__ vals, const float* __restrict__ carry,
    const float* __restrict__ xz, const float* __restrict__ decay,
    u16* __restrict__ ybf) {
  int idx = blockIdx.x * blockDim.x + threadIdx.x;       // (m,c)
  if (idx >= MROWS * D_INNER) return;
  int c = idx & (D_INNER - 1);
  int m = idx >> 11;
  int t = m & (TLEN - 1);
  int b = m >> 11;            // / TLEN
  int ch = t / CHUNK;
  int i  = t & (CHUNK - 1);

  float d = decay[c];
  float e = carry[((size_t)b * NCHUNK + ch) * D_INNER + c];
  float h = vals[idx] + __powf(d, (float)(i + 1)) * e;
  float z = xz[(size_t)m * (2 * D_INNER) + D_INNER + c];
  float y = h * silu_f(z);
  ybf[idx] = f32_to_bf16_rne(y);
}

// ---------------- host-side launcher ----------------
extern "C" void kernel_launch(void* const* d_in, const int* in_sizes, int n_in,
                              void* d_out, int out_size, void* d_ws, size_t ws_size,
                              hipStream_t stream) {
  const float* x      = (const float*)d_in[0];
  const float* W_in   = (const float*)d_in[1];
  const float* b_in   = (const float*)d_in[2];
  const float* W_conv = (const float*)d_in[3];
  const float* b_conv = (const float*)d_in[4];
  const float* W_dt   = (const float*)d_in[5];
  const float* b_dt   = (const float*)d_in[6];
  const float* W_out  = (const float*)d_in[7];
  const float* b_out  = (const float*)d_in[8];
  const float* decay  = (const float*)d_in[9];
  float* out          = (float*)d_out;

  // workspace carve-up (256B aligned)
  char* p = (char*)d_ws;
  auto carve = [&](size_t bytes) -> char* {
    char* r = p;
    p += (bytes + 255) & ~(size_t)255;
    return r;
  };
  u16*   x_bf    = (u16*)  carve((size_t)MROWS * D_MODEL * 2);        // 16 MB
  u16*   win_bf  = (u16*)  carve((size_t)2 * D_INNER * D_MODEL * 2);  //  8 MB
  u16*   wdt_bf  = (u16*)  carve((size_t)D_INNER * D_INNER * 2);      //  8 MB
  u16*   wout_bf = (u16*)  carve((size_t)D_MODEL * D_INNER * 2);      //  4 MB
  float* xz      = (float*)carve((size_t)MROWS * 2 * D_INNER * 4);    // 128 MB
  float* xconv_f = (float*)carve((size_t)MROWS * D_INNER * 4);        // 64 MB
  u16*   xconv_b = (u16*)  carve((size_t)MROWS * D_INNER * 2);        // 32 MB
  float* vals    = (float*)carve((size_t)MROWS * D_INNER * 4);        // 64 MB
  float* carry   = (float*)carve((size_t)BATCH * NCHUNK * D_INNER * 4);
  u16*   y_bf    = (u16*)  carve((size_t)MROWS * D_INNER * 2);        // 32 MB
  (void)ws_size; (void)in_sizes; (void)n_in; (void)out_size;

  const int TB = 256;
  auto blocks = [](int n, int tb) { return (n + tb - 1) / tb; };

  // 1) bf16 copies of x and weights
  cvt_f32_bf16_kernel<<<blocks(MROWS * D_MODEL, TB), TB, 0, stream>>>(
      x, x_bf, MROWS * D_MODEL);
  cvt_f32_bf16_kernel<<<blocks(2 * D_INNER * D_MODEL, TB), TB, 0, stream>>>(
      W_in, win_bf, 2 * D_INNER * D_MODEL);
  cvt_f32_bf16_kernel<<<blocks(D_INNER * D_INNER, TB), TB, 0, stream>>>(
      W_dt, wdt_bf, D_INNER * D_INNER);
  cvt_f32_bf16_kernel<<<blocks(D_MODEL * D_INNER, TB), TB, 0, stream>>>(
      W_out, wout_bf, D_MODEL * D_INNER);

  // 2) GEMM1: xz[8192,4096] = x_bf[8192,1024] @ W_in^T + b_in
  {
    dim3 grid((2 * D_INNER) / BN, MROWS / BM);
    gemm_bf16_wmma_kernel<0><<<grid, TB, 0, stream>>>(
        x_bf, win_bf, b_in, xz, nullptr, MROWS, 2 * D_INNER, D_MODEL);
  }

  // 3) causal depthwise conv + SiLU
  conv_silu_kernel<<<blocks(MROWS * D_INNER, TB), TB, 0, stream>>>(
      xz, W_conv, b_conv, xconv_f, xconv_b);

  // 4) GEMM2 + fused softplus*x_conv: vals[8192,2048]
  {
    dim3 grid(D_INNER / BN, MROWS / BM);
    gemm_bf16_wmma_kernel<2><<<grid, TB, 0, stream>>>(
        xconv_b, wdt_bf, b_dt, vals, xconv_f, MROWS, D_INNER, D_INNER);
  }

  // 5) chunked diagonal recurrence + gating -> y_bf
  scan_local_kernel<<<blocks(BATCH * NCHUNK * D_INNER, TB), TB, 0, stream>>>(
      vals, carry, decay);
  scan_carry_kernel<<<blocks(BATCH * D_INNER, TB), TB, 0, stream>>>(
      carry, decay);
  scan_fixup_gate_kernel<<<blocks(MROWS * D_INNER, TB), TB, 0, stream>>>(
      vals, carry, xz, decay, y_bf);

  // 6) GEMM3: out[8192,1024] = y_bf @ W_out^T + b_out
  {
    dim3 grid(D_MODEL / BN, MROWS / BM);
    gemm_bf16_wmma_kernel<0><<<grid, TB, 0, stream>>>(
        y_bf, wout_bf, b_out, out, nullptr, MROWS, D_MODEL, D_INNER);
  }
}
